// SentenceDecoder_25348896981015
// MI455X (gfx1250) — matrix operand order repaired and checked by reading, not compiled, not run on hardware
//
#include <hip/hip_runtime.h>
#include <hip/hip_bf16.h>

// ---------------------------------------------------------------------------
// CDNA5 (gfx1250) wave32 WMMA: D(16x16 f32) = A(16x32 bf16) x B(32x16 bf16) + C
// ---------------------------------------------------------------------------
typedef __attribute__((ext_vector_type(16))) __bf16 bf16x16;
typedef __attribute__((ext_vector_type(8)))  __bf16 bf16x8;
typedef __attribute__((ext_vector_type(4)))  __bf16 bf16x4;
typedef __attribute__((ext_vector_type(8)))  float  floatx8;

__device__ __forceinline__ __bf16 f2bf(float x) {
    union { float f; unsigned u; } c; c.f = x;
    unsigned r = (c.u + 0x7fffu + ((c.u >> 16) & 1u)) >> 16;   // RNE
    union { unsigned short s; __bf16 b; } o; o.s = (unsigned short)r;
    return o.b;
}

// A fragment (16x32 bf16): lane L row M=L&15, hi=L>>4;
// elems 0..7 -> K = k0+8*hi+i ; elems 8..15 -> K = k0+16+8*hi+(i-8).
__device__ __forceinline__ bf16x16 load_a_frag(const __bf16* __restrict__ A, int lda,
                                               int mt, int k0, int lane) {
    int hi = lane >> 4;
    const __bf16* base = A + (size_t)(mt * 16 + (lane & 15)) * lda + k0 + 8 * hi;
    union { bf16x16 v; bf16x8 h[2]; } u;
    u.h[0] = *(const bf16x8*)(base);
    u.h[1] = *(const bf16x8*)(base + 16);
    return u.v;
}

// B fragment (32x16 bf16) from W stored (N,K) row-major (B = W^T):
// lane L col N=L&15, hi=L>>4; elems 0..7 -> K = k0+8*hi+i ; elems 8..15 -> K = k0+16+8*hi+(i-8).
__device__ __forceinline__ bf16x16 load_b_frag(const __bf16* __restrict__ W, int ldw,
                                               int nt, int k0, int lane) {
    int hi = lane >> 4;
    const __bf16* base = W + (size_t)(nt * 16 + (lane & 15)) * ldw + k0 + 8 * hi;
    union { bf16x16 v; bf16x8 h[2]; } u;
    u.h[0] = *(const bf16x8*)(base);
    u.h[1] = *(const bf16x8*)(base + 16);
    return u.v;
}

__device__ __forceinline__ floatx8 wmma_bf16(bf16x16 a, bf16x16 b, floatx8 c) {
    return __builtin_amdgcn_wmma_f32_16x16x32_bf16(false, a, false, b, (short)0, c, false, false);
}

__device__ __forceinline__ float sigmoidf_(float x) { return 1.f / (1.f + __expf(-x)); }

// ---------------------------------------------------------------------------
// f32 -> bf16 bulk convert (4 elems/thread). n must be a multiple of 4.
// ---------------------------------------------------------------------------
__global__ void cvt_bf16_kernel(const float* __restrict__ src, __bf16* __restrict__ dst, int n4) {
    int i = blockIdx.x * 256 + threadIdx.x;
    if (i >= n4) return;
    float4 f = *(const float4*)(src + (size_t)i * 4);
    bf16x4 o;
    o[0] = f2bf(f.x); o[1] = f2bf(f.y); o[2] = f2bf(f.z); o[3] = f2bf(f.w);
    *(bf16x4*)(dst + (size_t)i * 4) = o;
}

// ---------------------------------------------------------------------------
// Generic GEMM: act( A[M,K](bf16) @ W[N,K]^T(bf16) + bias ) / rowdiv[row]
// -> optional f32 output and/or bf16 mirror (for feeding the next GEMM).
// Grid: x = N-tile-groups (8 waves/block, 16x64 per wave), y = M-tiles.
// Software-pipelined: fragment loads for iter k+1 issued before the 4 WMMAs
// of iter k, so the per-WMMA wait is partial (overlaps L2 latency).
// act: 0=none 1=relu 2=tanh.
// ---------------------------------------------------------------------------
__global__ void gemm_wmma(const __bf16* __restrict__ A, int lda,
                          const __bf16* __restrict__ W, int ldw,
                          const float* __restrict__ bias,
                          float* __restrict__ outf, __bf16* __restrict__ outbf, int ldc,
                          int N, int K, int act,
                          const float* __restrict__ rowdiv) {
    int wave = threadIdx.x >> 5, lane = threadIdx.x & 31;
    int numNT = N >> 4, numNT4 = (numNT + 3) >> 2;
    int nt4 = blockIdx.x * 8 + wave;
    int mt  = blockIdx.y;
    if (nt4 >= numNT4) return;                 // wave-uniform exit
    int nbase = nt4 * 4;
    bool ok1 = (nbase + 1) < numNT, ok2 = (nbase + 2) < numNT, ok3 = (nbase + 3) < numNT;
    int nt1 = ok1 ? nbase + 1 : numNT - 1;
    int nt2 = ok2 ? nbase + 2 : numNT - 1;
    int nt3 = ok3 ? nbase + 3 : numNT - 1;

    floatx8 acc0 = {}, acc1 = {}, acc2 = {}, acc3 = {};
    int wlane = lane & 15;

    // prologue: fragments for k0 = 0
    bf16x16 a  = load_a_frag(A, lda, mt, 0, lane);
    bf16x16 b0 = load_b_frag(W, ldw, nbase, 0, lane);
    bf16x16 b1 = load_b_frag(W, ldw, nt1,   0, lane);
    bf16x16 b2 = load_b_frag(W, ldw, nt2,   0, lane);
    bf16x16 b3 = load_b_frag(W, ldw, nt3,   0, lane);

    for (int k0 = 0; k0 < K; k0 += 32) {
        int kn = (k0 + 32 < K) ? k0 + 32 : k0;   // clamp on last iter (redundant reload)
        if (k0 + 64 < K) {   // prefetch 2 chunks ahead, keep near (global_prefetch_b8)
            __builtin_prefetch((const void*)(W + (size_t)(nbase * 16 + wlane) * ldw + k0 + 64), 0, 3);
            __builtin_prefetch((const void*)(W + (size_t)(nt3 * 16 + wlane) * ldw + k0 + 64), 0, 3);
        }
        // issue next-iteration loads first
        bf16x16 an  = load_a_frag(A, lda, mt, kn, lane);
        bf16x16 bn0 = load_b_frag(W, ldw, nbase, kn, lane);
        bf16x16 bn1 = load_b_frag(W, ldw, nt1,   kn, lane);
        bf16x16 bn2 = load_b_frag(W, ldw, nt2,   kn, lane);
        bf16x16 bn3 = load_b_frag(W, ldw, nt3,   kn, lane);
        // compute on current fragments
        acc0 = wmma_bf16(a, b0, acc0);
        acc1 = wmma_bf16(a, b1, acc1);
        acc2 = wmma_bf16(a, b2, acc2);
        acc3 = wmma_bf16(a, b3, acc3);
        a = an; b0 = bn0; b1 = bn1; b2 = bn2; b3 = bn3;
    }

    int hi = lane >> 4, nn = lane & 15;
    int baserow = mt * 16;
#define STORE_TILE(ACC, NT, VALID)                                              \
    if (VALID) {                                                                \
        int col = (NT) * 16 + nn;                                               \
        float bv = bias ? bias[col] : 0.f;                                      \
        _Pragma("unroll")                                                       \
        for (int r = 0; r < 8; r++) {                                           \
            int row = baserow + r + 8 * hi;                                     \
            float val = ACC[r] + bv;                                            \
            if (act == 1) val = fmaxf(val, 0.f);                                \
            else if (act == 2) val = tanhf(val);                                \
            if (rowdiv) val /= rowdiv[row];                                     \
            if (outf)  outf[(size_t)row * ldc + col] = val;                     \
            if (outbf) outbf[(size_t)row * ldc + col] = f2bf(val);              \
        }                                                                       \
    }
    STORE_TILE(acc0, nbase,     true)
    STORE_TILE(acc1, nbase + 1, ok1)
    STORE_TILE(acc2, nbase + 2, ok2)
    STORE_TILE(acc3, nbase + 3, ok3)
#undef STORE_TILE
}

// ---------------------------------------------------------------------------
// LSTM cell: gates = gate_x[t] + h @ w_hh^T + b_hh ; elementwise fused.
// Grid dim3(4,4), 256 threads: blockIdx.y = mt (batch tile), wave -> nt.
// Same software pipelining as gemm_wmma. h in/out bf16; m stays f32.
// ---------------------------------------------------------------------------
__global__ void lstm_cell_wmma(const __bf16* __restrict__ h_in, const float* __restrict__ m_in,
                               const __bf16* __restrict__ w_hh, const float* __restrict__ gx,
                               const float* __restrict__ b_hh,
                               __bf16* __restrict__ h_out, float* __restrict__ m_out) {
    int wave = threadIdx.x >> 5, lane = threadIdx.x & 31;
    int nt = blockIdx.x * 8 + wave;   // 0..31 (hidden col tile)
    int mt = blockIdx.y;              // 0..3  (batch tile)

    floatx8 acc_i = {}, acc_f = {}, acc_g = {}, acc_o = {};
    bf16x16 a  = load_a_frag(h_in, 512, mt, 0, lane);
    bf16x16 bi = load_b_frag(w_hh, 512, nt,      0, lane);
    bf16x16 bf = load_b_frag(w_hh, 512, 32 + nt, 0, lane);
    bf16x16 bg = load_b_frag(w_hh, 512, 64 + nt, 0, lane);
    bf16x16 bo = load_b_frag(w_hh, 512, 96 + nt, 0, lane);
    for (int k0 = 0; k0 < 512; k0 += 32) {
        int kn = (k0 + 32 < 512) ? k0 + 32 : k0;
        bf16x16 an  = load_a_frag(h_in, 512, mt, kn, lane);
        bf16x16 bni = load_b_frag(w_hh, 512, nt,      kn, lane);
        bf16x16 bnf = load_b_frag(w_hh, 512, 32 + nt, kn, lane);
        bf16x16 bng = load_b_frag(w_hh, 512, 64 + nt, kn, lane);
        bf16x16 bno = load_b_frag(w_hh, 512, 96 + nt, kn, lane);
        acc_i = wmma_bf16(a, bi, acc_i);
        acc_f = wmma_bf16(a, bf, acc_f);
        acc_g = wmma_bf16(a, bg, acc_g);
        acc_o = wmma_bf16(a, bo, acc_o);
        a = an; bi = bni; bf = bnf; bg = bng; bo = bno;
    }
    int hi = lane >> 4, nn = lane & 15;
    int j = nt * 16 + nn;
    float bhi = b_hh[j], bhf = b_hh[512 + j], bhg = b_hh[1024 + j], bho = b_hh[1536 + j];
#pragma unroll
    for (int r = 0; r < 8; r++) {
        int b = mt * 16 + r + 8 * hi;
        const float* grow = gx + (size_t)b * 2048;
        float gi = acc_i[r] + grow[j]        + bhi;
        float gf = acc_f[r] + grow[512 + j]  + bhf;
        float gg = acc_g[r] + grow[1024 + j] + bhg;
        float go = acc_o[r] + grow[1536 + j] + bho;
        float mn = sigmoidf_(gf) * m_in[(size_t)b * 512 + j] + sigmoidf_(gi) * tanhf(gg);
        float hn = sigmoidf_(go) * tanhf(mn);
        m_out[(size_t)b * 512 + j] = mn;
        h_out[(size_t)b * 512 + j] = f2bf(hn);
    }
}

// ---------------------------------------------------------------------------
// image mean over P=64; writes f32 (buildx use) + bf16 (GEMM A operand).
// ---------------------------------------------------------------------------
__global__ void mean_kernel(const float* __restrict__ image,
                            float* __restrict__ img_mean, __bf16* __restrict__ img_mean_bf) {
    int idx = blockIdx.x * 256 + threadIdx.x;      // b*512 + e
    int b = idx >> 9, e = idx & 511;
    float s = 0.f;
#pragma unroll 8
    for (int p = 0; p < 64; p++) s += image[((size_t)(b * 64 + p)) * 512 + e];
    s *= (1.f / 64.f);
    img_mean[idx] = s;
    img_mean_bf[idx] = f2bf(s);
}

// ---------------------------------------------------------------------------
// Build X[T*B, 1568] (bf16, GEMM A operand) =
//   [img_mean | view_position | label | topic | emb(tok)]
// ---------------------------------------------------------------------------
__global__ void buildx_kernel(const float* __restrict__ img_mean,
                              const float* __restrict__ view_position,
                              const float* __restrict__ label,
                              const float* __restrict__ topic,
                              const float* __restrict__ word_emb,
                              const int* __restrict__ text,
                              __bf16* __restrict__ X) {
    int r = blockIdx.x;                 // r = t*64 + b
    int t = r >> 6, b = r & 63;
    int tok = (t == 0) ? 0 : text[b * 32 + (t - 1)];
    __bf16* xr = X + (size_t)r * 1568;
    for (int i = threadIdx.x; i < 1568; i += 256) {
        float val;
        if (i < 512)        val = img_mean[b * 512 + i];
        else if (i < 528)   val = view_position[b * 16 + (i - 512)];
        else if (i < 544)   val = label[b * 16 + (i - 528)];
        else if (i < 1056)  val = topic[b * 512 + (i - 544)];
        else                val = word_emb[(size_t)tok * 512 + (i - 1056)];
        xr[i] = f2bf(val);
    }
}

// ---------------------------------------------------------------------------
// Attention: z = <tanh(ext + _hh), fc_z_w> ; a=softmax(z) ; c = a @ feats ;
// chh = c + hh (bf16, feeds vocab GEMM). One block per batch row.
// ---------------------------------------------------------------------------
__global__ void attn_kernel(const float* __restrict__ v, const float* __restrict__ hh_,
                            const float* __restrict__ s_, const float* __restrict__ fc_z_w,
                            const float* __restrict__ fc_z_b,
                            const float* __restrict__ image, const float* __restrict__ hs,
                            float* __restrict__ attn_out, int t, __bf16* __restrict__ chh_bf) {
    __shared__ float shh[512];
    __shared__ float szw[512];
    __shared__ float sz[65];
    int b = blockIdx.x, tid = threadIdx.x;
    for (int i = tid; i < 512; i += 256) { shh[i] = hh_[b * 512 + i]; szw[i] = fc_z_w[i]; }
    __syncthreads();
    int wave = tid >> 5, lane = tid & 31;
    for (int p = wave; p < 65; p += 8) {
        const float* src = (p < 64) ? (v + ((size_t)(b * 64 + p)) * 512) : (s_ + (size_t)b * 512);
        float partial = 0.f;
        for (int h = lane; h < 512; h += 32)
            partial += tanhf(src[h] + shh[h]) * szw[h];
        for (int off = 16; off > 0; off >>= 1) partial += __shfl_down(partial, off, 32);
        if (lane == 0) sz[p] = partial + fc_z_b[0];
    }
    __syncthreads();
    if (tid == 0) {
        float mx = sz[0];
        for (int p = 1; p < 65; p++) mx = fmaxf(mx, sz[p]);
        float sum = 0.f;
        for (int p = 0; p < 65; p++) { float e = __expf(sz[p] - mx); sz[p] = e; sum += e; }
        float inv = 1.f / sum;
        for (int p = 0; p < 65; p++) sz[p] *= inv;
    }
    __syncthreads();
    if (tid < 65) attn_out[(size_t)b * 32 * 65 + (size_t)t * 65 + tid] = sz[tid];
    for (int e = tid; e < 512; e += 256) {
        float cc = 0.f;
#pragma unroll 4
        for (int p = 0; p < 64; p++) cc += sz[p] * image[((size_t)(b * 64 + p)) * 512 + e];
        cc += sz[64] * hs[(size_t)b * 1024 + 512 + e];                   // s row of feats
        chh_bf[(size_t)b * 512 + e] = f2bf(cc + hs[(size_t)b * 1024 + e]); // + hh
    }
}

// ---------------------------------------------------------------------------
// log-softmax stats per row over V=30000: lse[b]; accumulate sum_log_prob.
// ---------------------------------------------------------------------------
__global__ void lse_kernel(float* __restrict__ logp, int t,
                           float* __restrict__ lse, float* __restrict__ slp) {
    __shared__ float red[256];
    int b = blockIdx.x, tid = threadIdx.x;
    float* row = logp + ((size_t)b * 32 + t) * 30000;
    float mx = -1e30f;
    for (int i = tid; i < 30000; i += 256) mx = fmaxf(mx, row[i]);
    red[tid] = mx; __syncthreads();
    for (int s = 128; s > 0; s >>= 1) { if (tid < s) red[tid] = fmaxf(red[tid], red[tid + s]); __syncthreads(); }
    mx = red[0]; __syncthreads();
    float sum = 0.f;
    for (int i = tid; i < 30000; i += 256) sum += __expf(row[i] - mx);
    red[tid] = sum; __syncthreads();
    for (int s = 128; s > 0; s >>= 1) { if (tid < s) red[tid] += red[tid + s]; __syncthreads(); }
    if (tid == 0) {
        float l = mx + logf(red[0]);
        lse[b] = l;
        float prev = (t == 0) ? 0.f : slp[b];
        slp[b] = prev + (mx - l);          // max_v logp = max_logit - lse
    }
}

__global__ void sub_kernel(float* __restrict__ logp, int t, const float* __restrict__ lse) {
    int idx = blockIdx.x * 256 + threadIdx.x;      // 64*30000 total
    int b = idx / 30000, vv = idx - b * 30000;
    logp[((size_t)b * 32 + t) * 30000 + vv] -= lse[b];
}

// ---------------------------------------------------------------------------
// Host driver
// ---------------------------------------------------------------------------
static inline dim3 gemm_grid(int M, int N) {
    int numNT4 = ((N >> 4) + 3) >> 2;
    return dim3((unsigned)((numNT4 + 7) / 8), (unsigned)(M >> 4));
}
static inline void cvt_launch(const float* src, __bf16* dst, size_t n, hipStream_t s) {
    int n4 = (int)(n / 4);
    cvt_bf16_kernel<<<(n4 + 255) / 256, 256, 0, s>>>(src, dst, n4);
}

extern "C" void kernel_launch(void* const* d_in, const int* in_sizes, int n_in,
                              void* d_out, int out_size, void* d_ws, size_t ws_size,
                              hipStream_t stream) {
    const float* image         = (const float*)d_in[0];
    const float* view_position = (const float*)d_in[1];
    const float* label         = (const float*)d_in[2];
    const float* topic         = (const float*)d_in[3];
    const float* temp          = (const float*)d_in[4];
    const int*   text          = (const int*)  d_in[5];
    const float* word_emb      = (const float*)d_in[6];
    const float* fc_v_w        = (const float*)d_in[7];
    const float* fc_v_b        = (const float*)d_in[8];
    const float* fc_h_w        = (const float*)d_in[9];
    const float* fc_h_b        = (const float*)d_in[10];
    const float* fc_m_w        = (const float*)d_in[11];
    const float* fc_m_b        = (const float*)d_in[12];
    const float* w_ih          = (const float*)d_in[13];
    const float* w_hh          = (const float*)d_in[14];
    const float* b_ih          = (const float*)d_in[15];
    const float* b_hh          = (const float*)d_in[16];
    const float* fc_w          = (const float*)d_in[17];
    const float* fc_b          = (const float*)d_in[18];
    const float* fc_hh_w       = (const float*)d_in[19];
    const float* fc_hh_b       = (const float*)d_in[20];
    const float* fc_s_w        = (const float*)d_in[21];
    const float* fc_s_b        = (const float*)d_in[22];
    const float* fc_z_w        = (const float*)d_in[23];
    const float* fc_z_b        = (const float*)d_in[24];
    const float* fc_p_w        = (const float*)d_in[25];
    const float* fc_p_b        = (const float*)d_in[26];

    // ---- workspace carve-up (256B aligned) ----
    char* wp = (char*)d_ws;
    auto alloc = [&](size_t bytes) -> void* {
        void* r = (void*)wp; wp += (bytes + 255) & ~(size_t)255; return r;
    };
    float*  img_mean    = (float*) alloc(64 * 512 * 4);
    __bf16* img_mean_bf = (__bf16*)alloc(64 * 512 * 2);
    float*  vbuf        = (float*) alloc((size_t)4096 * 512 * 4);
    __bf16* Xbf         = (__bf16*)alloc((size_t)2048 * 1568 * 2);
    float*  gatex       = (float*) alloc((size_t)32 * 64 * 2048 * 4);
    __bf16* hbf0        = (__bf16*)alloc(64 * 512 * 2);
    __bf16* hbf1        = (__bf16*)alloc(64 * 512 * 2);
    float*  mbuf0       = (float*) alloc(64 * 512 * 4);
    float*  mbuf1       = (float*) alloc(64 * 512 * 4);
    float*  hsbuf       = (float*) alloc(64 * 1024 * 4);
    __bf16* hsbf        = (__bf16*)alloc(64 * 1024 * 2);
    float*  hhbuf       = (float*) alloc(64 * 512 * 4);
    float*  sbuf        = (float*) alloc(64 * 512 * 4);
    __bf16* chhbf       = (__bf16*)alloc(64 * 512 * 2);
    float*  lsebuf      = (float*) alloc(64 * 4);
    __bf16* image_bf    = (__bf16*)alloc((size_t)4096 * 512 * 2);
    __bf16* fc_v_w_bf   = (__bf16*)alloc((size_t)512 * 512 * 2);
    __bf16* fc_h_w_bf   = (__bf16*)alloc((size_t)512 * 512 * 2);
    __bf16* fc_m_w_bf   = (__bf16*)alloc((size_t)512 * 512 * 2);
    __bf16* w_ih_bf     = (__bf16*)alloc((size_t)2048 * 1568 * 2);
    __bf16* w_hh_bf     = (__bf16*)alloc((size_t)2048 * 512 * 2);
    __bf16* fc_w_bf     = (__bf16*)alloc((size_t)1024 * 512 * 2);
    __bf16* fc_hh_w_bf  = (__bf16*)alloc((size_t)512 * 512 * 2);
    __bf16* fc_s_w_bf   = (__bf16*)alloc((size_t)512 * 512 * 2);
    __bf16* fc_p_w_bf   = (__bf16*)alloc((size_t)30000 * 512 * 2);

    float* out  = (float*)d_out;
    float* logp = out;                                   // (B,T,V)
    float* attn = out + (size_t)64 * 32 * 30000;         // (B,T,65)
    float* slp  = attn + (size_t)64 * 32 * 65;           // (B,)

    // ---- one-time conversions (HBM-cheap, amortized over 32 steps) ----
    cvt_launch(image,   image_bf,   (size_t)4096 * 512,  stream);
    cvt_launch(fc_v_w,  fc_v_w_bf,  (size_t)512 * 512,   stream);
    cvt_launch(fc_h_w,  fc_h_w_bf,  (size_t)512 * 512,   stream);
    cvt_launch(fc_m_w,  fc_m_w_bf,  (size_t)512 * 512,   stream);
    cvt_launch(w_ih,    w_ih_bf,    (size_t)2048 * 1568, stream);
    cvt_launch(w_hh,    w_hh_bf,    (size_t)2048 * 512,  stream);
    cvt_launch(fc_w,    fc_w_bf,    (size_t)1024 * 512,  stream);
    cvt_launch(fc_hh_w, fc_hh_w_bf, (size_t)512 * 512,   stream);
    cvt_launch(fc_s_w,  fc_s_w_bf,  (size_t)512 * 512,   stream);
    cvt_launch(fc_p_w,  fc_p_w_bf,  (size_t)30000 * 512, stream);

    // ---- step-independent precompute ----
    mean_kernel<<<128, 256, 0, stream>>>(image, img_mean, img_mean_bf);
    gemm_wmma<<<gemm_grid(4096, 512), 256, 0, stream>>>(
        image_bf, 512, fc_v_w_bf, 512, fc_v_b, vbuf, nullptr, 512, 512, 512, 0, nullptr);
    gemm_wmma<<<gemm_grid(64, 512), 256, 0, stream>>>(
        img_mean_bf, 512, fc_h_w_bf, 512, fc_h_b, nullptr, hbf0, 512, 512, 512, 2, nullptr);
    gemm_wmma<<<gemm_grid(64, 512), 256, 0, stream>>>(
        img_mean_bf, 512, fc_m_w_bf, 512, fc_m_b, mbuf0, nullptr, 512, 512, 512, 2, nullptr);
    buildx_kernel<<<2048, 256, 0, stream>>>(img_mean, view_position, label, topic,
                                            word_emb, text, Xbf);
    gemm_wmma<<<gemm_grid(2048, 2048), 256, 0, stream>>>(
        Xbf, 1568, w_ih_bf, 1568, b_ih, gatex, nullptr, 2048, 2048, 1568, 0, nullptr);

    // ---- sequential decode ----
    __bf16* hc = hbf0; __bf16* hn = hbf1;
    float*  mc = mbuf0; float* mn = mbuf1;
    for (int t = 0; t < 32; t++) {
        lstm_cell_wmma<<<dim3(4, 4), 256, 0, stream>>>(hc, mc, w_hh_bf,
                                                       gatex + (size_t)t * 64 * 2048,
                                                       b_hh, hn, mn);
        gemm_wmma<<<gemm_grid(64, 1024), 256, 0, stream>>>(
            hn, 512, fc_w_bf, 512, fc_b, hsbuf, hsbf, 1024, 1024, 512, 1, nullptr);
        gemm_wmma<<<gemm_grid(64, 512), 256, 0, stream>>>(
            hsbf, 1024, fc_hh_w_bf, 512, fc_hh_b, hhbuf, nullptr, 512, 512, 512, 0, nullptr);
        gemm_wmma<<<gemm_grid(64, 512), 256, 0, stream>>>(
            hsbf + 512, 1024, fc_s_w_bf, 512, fc_s_b, sbuf, nullptr, 512, 512, 512, 0, nullptr);
        attn_kernel<<<64, 256, 0, stream>>>(vbuf, hhbuf, sbuf, fc_z_w, fc_z_b,
                                            image, hsbuf, attn, t, chhbf);
        gemm_wmma<<<gemm_grid(64, 30000), 256, 0, stream>>>(
            chhbf, 512, fc_p_w_bf, 512, fc_p_b, logp + (size_t)t * 30000, nullptr,
            32 * 30000, 30000, 512, 0, temp);
        lse_kernel<<<64, 256, 0, stream>>>(logp, t, lsebuf, slp);
        sub_kernel<<<7500, 256, 0, stream>>>(logp, t, lsebuf);
        __bf16* th = hc; hc = hn; hn = th;
        float*  tm = mc; mc = mn; mn = tm;
    }
    (void)in_sizes; (void)n_in; (void)out_size; (void)ws_size;
}
